// DegreeAwareDynamicGNN_21328807592522
// MI455X (gfx1250) — matrix operand order repaired and checked by reading, not compiled, not run on hardware
//
#include <hip/hip_runtime.h>
#include <math.h>

// ---------------------------------------------------------------------------
// CDNA5 (gfx1250) GCN pipeline.
//  - GEMMs: v_wmma_f32_16x16x32_bf16, B-panel staged in LDS via the gfx1250
//    async global->LDS path (ASYNCcnt) when available, CT (8/4/3) accumulators/wave.
//  - Scatter: one thread per (edge, 4-feature chunk) -> b128 payload loads,
//    4x fewer index loads, global_atomic_add_f32.
// ---------------------------------------------------------------------------

typedef __attribute__((ext_vector_type(16))) __bf16 v16bf;
typedef __attribute__((ext_vector_type(8)))  float  v8f;
typedef long long i64;

#define TPB 256

#if defined(__has_builtin)
#if __has_builtin(__builtin_amdgcn_global_load_async_to_lds_b128) && \
    __has_builtin(__builtin_amdgcn_s_wait_asynccnt)
#define ASYNC_LDS 1
#endif
#endif

#ifdef ASYNC_LDS
// Builtin signature (from clang diagnostic): param0 is
//   '__attribute__((vector_size(16))) int __device__ *'  (addrspace(1))
// param1 symmetrically the LDS (addrspace(3)) pointer.
typedef int v4i __attribute__((vector_size(16)));
typedef v4i __attribute__((address_space(1)))* as1_v4i_p;
typedef v4i __attribute__((address_space(3)))* as3_v4i_p;
#endif

static inline unsigned grid1d(long n) { return (unsigned)((n + TPB - 1) / TPB); }

// ---- degree accumulation (out-degree of src, in-degree of dst) ------------
__global__ void k_edge_deg(const i64* __restrict__ src, const i64* __restrict__ dst,
                           float* __restrict__ deg_src, float* __restrict__ deg_dst,
                           long E) {
  long e = blockIdx.x * (long)blockDim.x + threadIdx.x;
  if (e >= E) return;
  atomicAdd(deg_src + (int)src[e], 1.0f);
  atomicAdd(deg_dst + (int)dst[e], 1.0f);
}

// ---- max degree via uint-bit atomicMax (valid for non-negative floats) ----
__global__ void k_maxdeg(const float* __restrict__ deg, unsigned* __restrict__ maxbits, int N) {
  int i = blockIdx.x * blockDim.x + threadIdx.x;
  if (i >= N) return;
  atomicMax(maxbits, __float_as_uint(deg[i]));
}

// ---- per-node scalars: nd, dinv, dw, s1=param1*dw, s2=param2*dw -----------
__global__ void k_nodescalar(const float* __restrict__ deg_src, const float* __restrict__ deg_dst,
                             const unsigned* __restrict__ maxbits,
                             const float* __restrict__ param1, const float* __restrict__ param2,
                             const float* __restrict__ deg_W, const float* __restrict__ deg_b,
                             const float* __restrict__ att_W, const float* __restrict__ att_b,
                             float* __restrict__ nd, float* __restrict__ dinv,
                             float* __restrict__ s1, float* __restrict__ s2,
                             int N, int H) {
  int i = blockIdx.x * blockDim.x + threadIdx.x;
  if (i >= N) return;
  float mx = __uint_as_float(*maxbits);
  if (!(mx > 0.0f)) mx = 1.0f;
  float ndv = deg_src[i] / mx;
  float z = att_b[0];
  for (int h = 0; h < H; ++h) z += (ndv * deg_W[h] + deg_b[h]) * att_W[h];
  float dw = 1.0f / (1.0f + __expf(-z));
  nd[i]   = ndv;
  dinv[i] = rsqrtf(deg_dst[i] + 1.0f);   // self-loop degree
  s1[i]   = param1[i] * dw;
  s2[i]   = param2[i] * dw;
}

// ---- build scaled xa = [x | demb] * s1, converted to bf16 -----------------
__global__ void k_build_xa(const float* __restrict__ x, const float* __restrict__ nd,
                           const float* __restrict__ s1,
                           const float* __restrict__ deg_W, const float* __restrict__ deg_b,
                           __bf16* __restrict__ xa, int N, int F, int H) {
  int K = F + H;
  long idx = blockIdx.x * (long)blockDim.x + threadIdx.x;
  if (idx >= (long)N * K) return;
  int i = (int)(idx / K);
  int f = (int)(idx % K);
  float v = (f < F) ? x[(long)i * F + f] : (nd[i] * deg_W[f - F] + deg_b[f - F]);
  xa[idx] = (__bf16)(v * s1[i]);
}

// ---- pack weights: W[k, n] (row-major [K, n_orig]) -> Wt[n][k] bf16, pad n
__global__ void k_pack_weight(const float* __restrict__ W, __bf16* __restrict__ Wt,
                              int K, int n_orig, int n_pad) {
  long idx = blockIdx.x * (long)blockDim.x + threadIdx.x;
  if (idx >= (long)K * n_pad) return;
  int n = (int)(idx / K);
  int k = (int)(idx % K);
  Wt[(long)n * K + k] = (n < n_orig) ? (__bf16)W[(long)k * n_orig + n] : (__bf16)0.0f;
}

// ---- WMMA GEMM with LDS-staged B panel ------------------------------------
// C[nrows, ncols] = A[nrows, K] * Bt[ncols, K]^T,  ncols == CT*16, K % 32 == 0.
// Block = 256 threads = 8 waves; each wave owns one 16-row tile and keeps CT
// accumulators so every A fragment feeds CT back-to-back WMMAs.
// B panel staged in LDS once per block (async global->LDS when available),
// row pitch padded by 8 bf16 to spread DS banks.
template <int CT>
__global__ void k_wmma_gemm_lds(const __bf16* __restrict__ A, const __bf16* __restrict__ Bt,
                                float* __restrict__ C, int nrows, int K, int ncols) {
  extern __shared__ __bf16 sB[];          // [ncols][K+8]
  const int Kp = K + 8;

  // ---- stage B panel: 16B chunks, strided across the block ----
  int chunksPerRow = K >> 3;              // 8 bf16 per 16B chunk
  int totalChunks  = ncols * chunksPerRow;
  for (int c = threadIdx.x; c < totalChunks; c += blockDim.x) {
    int n  = c / chunksPerRow;
    int kc = c - n * chunksPerRow;
    const __bf16* gsrc = Bt + (long)n * K + (kc << 3);
    __bf16*       ldst = sB + (long)n * Kp + (kc << 3);
#ifdef ASYNC_LDS
    __builtin_amdgcn_global_load_async_to_lds_b128(
        (as1_v4i_p)gsrc, (as3_v4i_p)ldst, 0, 0);
#else
    *(uint4*)ldst = *(const uint4*)gsrc;
#endif
  }
#ifdef ASYNC_LDS
  __builtin_amdgcn_s_wait_asynccnt(0);
#endif
  __syncthreads();

  // ---- per-wave 16-row tile ----
  int wave = threadIdx.x >> 5;
  int lane = threadIdx.x & 31;
  int m0   = ((blockIdx.x << 3) + wave) << 4;
  if (m0 >= nrows) return;                // wave-uniform; EXEC all-ones below
  int half = lane >> 4, l16 = lane & 15;
  int arow = m0 + l16;
  if (arow >= nrows) arow = nrows - 1;    // clamp (stores are guarded)
  const __bf16* ab = A + (long)arow * K;

  v8f acc[CT] = {};
  for (int k0 = 0; k0 < K; k0 += 32) {
    union { v16bf v; uint4 q[2]; } a;
    a.q[0] = *(const uint4*)(ab + k0 + 8 * half);
    a.q[1] = *(const uint4*)(ab + k0 + 16 + 8 * half);
#pragma unroll
    for (int t = 0; t < CT; ++t) {
      union { v16bf v; uint4 q[2]; } b;
      const __bf16* bl = sB + (long)((t << 4) + l16) * Kp + k0 + 8 * half;
      b.q[0] = *(const uint4*)(bl);
      b.q[1] = *(const uint4*)(bl + 16);
      acc[t] = __builtin_amdgcn_wmma_f32_16x16x32_bf16(
          /*neg_a=*/false, a.v, /*neg_b=*/false, b.v,
          /*c_mod=*/(short)0, acc[t], /*reuse_a=*/false, /*reuse_b=*/false);
    }
  }

#pragma unroll
  for (int t = 0; t < CT; ++t) {
    int col = (t << 4) + l16;
#pragma unroll
    for (int r = 0; r < 8; ++r) {
      int row = m0 + r + 8 * half;
      if (row < nrows) C[(long)row * ncols + col] = acc[t][r];
    }
  }
}

// ---- self-loop + bias init: out[i,f] = b[f] + h[i,f] * dinv[i]^2 ----------
__global__ void k_selfinit(const float* __restrict__ h, const float* __restrict__ bias,
                           const float* __restrict__ dinv, float* __restrict__ out,
                           int N, int dim, int ldh) {
  long idx = blockIdx.x * (long)blockDim.x + threadIdx.x;
  if (idx >= (long)N * dim) return;
  int i = (int)(idx / dim);
  int f = (int)(idx % dim);
  float dv = dinv[i];
  out[idx] = bias[f] + h[(long)i * ldh + f] * dv * dv;
}

// ---- edge scatter: out[dst] += h[src] * dinv[src]*dinv[dst] ---------------
// One thread per (edge, 4-feature chunk): b128 payload, indices loaded once.
__global__ void k_scatter4(const float* __restrict__ h, const i64* __restrict__ src,
                           const i64* __restrict__ dst, const float* __restrict__ dinv,
                           float* __restrict__ out, long E, int dim4, int ldh, int ldo) {
  long idx = blockIdx.x * (long)blockDim.x + threadIdx.x;
  if (idx >= E * dim4) return;
  long e = idx / dim4;
  int  c = (int)(idx - e * dim4) << 2;
  int s = (int)src[e], d = (int)dst[e];
  float w = dinv[s] * dinv[d];
  float4 v = *(const float4*)(h + (long)s * ldh + c);
  float* o = out + (long)d * ldo + c;
  atomicAdd(o + 0, v.x * w);
  atomicAdd(o + 1, v.y * w);
  atomicAdd(o + 2, v.z * w);
  atomicAdd(o + 3, v.w * w);
}

// ---- post layer 1: x1 = relu(gcn1) * s2; bf16 copy + fp32 residual half ---
__global__ void k_post1(const float* __restrict__ gcn1, const float* __restrict__ s2,
                        __bf16* __restrict__ x1b, float* __restrict__ x1res,
                        int N, int dim, int resdim) {
  long idx = blockIdx.x * (long)blockDim.x + threadIdx.x;
  if (idx >= (long)N * dim) return;
  int i = (int)(idx / dim);
  int f = (int)(idx % dim);
  float v = gcn1[idx];
  v = (v > 0.0f ? v : 0.0f) * s2[i];
  x1b[idx] = (__bf16)v;
  if (f < resdim) x1res[(long)i * resdim + f] = v;
}

// ---- post layer 2: x2 = relu(gcn2) + x1[:, :H]; bf16 ----------------------
__global__ void k_post2(const float* __restrict__ gcn2, const float* __restrict__ x1res,
                        __bf16* __restrict__ x2b, int N, int dim) {
  long idx = blockIdx.x * (long)blockDim.x + threadIdx.x;
  if (idx >= (long)N * dim) return;
  float v = gcn2[idx];
  v = (v > 0.0f ? v : 0.0f) + x1res[idx];
  x2b[idx] = (__bf16)v;
}

// ---- log-softmax over C classes per node ----------------------------------
__global__ void k_logsoftmax(const float* __restrict__ gcn3, float* __restrict__ out,
                             int N, int C) {
  int i = blockIdx.x * blockDim.x + threadIdx.x;
  if (i >= N) return;
  const float* p = gcn3 + (long)i * C;
  float m = p[0];
  for (int c = 1; c < C; ++c) m = fmaxf(m, p[c]);
  float s = 0.0f;
  for (int c = 0; c < C; ++c) s += __expf(p[c] - m);
  float l = __logf(s);
  float* o = out + (long)i * C;
  for (int c = 0; c < C; ++c) o[c] = p[c] - m - l;
}

// ---------------------------------------------------------------------------

extern "C" void kernel_launch(void* const* d_in, const int* in_sizes, int n_in,
                              void* d_out, int out_size, void* d_ws, size_t ws_size,
                              hipStream_t stream) {
  // Inputs (setup_inputs order)
  const float* x      = (const float*)d_in[0];
  const i64*   eidx   = (const i64*)  d_in[1];
  const float* param1 = (const float*)d_in[2];
  const float* param2 = (const float*)d_in[3];
  const float* deg_W  = (const float*)d_in[4];
  const float* deg_b  = (const float*)d_in[5];
  const float* att_W  = (const float*)d_in[6];
  const float* att_b  = (const float*)d_in[7];
  const float* W1     = (const float*)d_in[8];
  const float* b1     = (const float*)d_in[9];
  const float* W2     = (const float*)d_in[10];
  const float* b2     = (const float*)d_in[11];
  const float* W3     = (const float*)d_in[12];
  const float* b3     = (const float*)d_in[13];

  const int  N  = in_sizes[2];            // 100000
  const long E  = in_sizes[1] / 2;        // 1.6M
  const int  F  = in_sizes[0] / N;        // 128
  const int  H  = in_sizes[5];            // 64
  const int  H2 = in_sizes[9];            // 128 (=2H)
  const int  C  = in_sizes[13];           // 40
  const int  Cp = (C + 15) & ~15;         // 48 (WMMA column padding)
  const int  K1 = F + H;                  // 192

  const i64* src = eidx;
  const i64* dst = eidx + E;

  // Workspace carve-out (256B aligned)
  char*  w   = (char*)d_ws;
  size_t off = 0;
  auto alloc = [&](size_t bytes) -> void* {
    void* p = w + off;
    off = (off + bytes + 255) & ~(size_t)255;
    return p;
  };

  float*    deg_src = (float*)   alloc((size_t)N * 4);
  float*    deg_dst = (float*)   alloc((size_t)N * 4);
  unsigned* maxbits = (unsigned*)alloc(256);
  float*    nd      = (float*)   alloc((size_t)N * 4);
  float*    dinv    = (float*)   alloc((size_t)N * 4);
  float*    s1      = (float*)   alloc((size_t)N * 4);
  float*    s2      = (float*)   alloc((size_t)N * 4);
  __bf16*   xa      = (__bf16*)  alloc((size_t)N * K1 * 2);
  __bf16*   W1t     = (__bf16*)  alloc((size_t)H2 * K1 * 2);
  __bf16*   W2t     = (__bf16*)  alloc((size_t)H  * H2 * 2);
  __bf16*   W3t     = (__bf16*)  alloc((size_t)Cp * H  * 2);
  float*    h1      = (float*)   alloc((size_t)N * H2 * 4);
  float*    g1      = (float*)   alloc((size_t)N * H2 * 4);
  __bf16*   x1b     = (__bf16*)  alloc((size_t)N * H2 * 2);
  float*    x1res   = (float*)   alloc((size_t)N * H  * 4);
  float*    h2      = (float*)   alloc((size_t)N * H  * 4);
  float*    g2      = (float*)   alloc((size_t)N * H  * 4);
  __bf16*   x2b     = (__bf16*)  alloc((size_t)N * H  * 2);
  float*    h3      = (float*)   alloc((size_t)N * Cp * 4);
  float*    g3      = (float*)   alloc((size_t)N * C  * 4);
  (void)ws_size; (void)n_in; (void)out_size;

  const int rowTiles   = (N + 15) / 16;
  const int gemmBlocks = (rowTiles + 7) / 8;

  // --- degrees ---
  (void)hipMemsetAsync(deg_src, 0, (size_t)N * 4, stream);
  (void)hipMemsetAsync(deg_dst, 0, (size_t)N * 4, stream);
  (void)hipMemsetAsync(maxbits, 0, 4, stream);
  k_edge_deg<<<grid1d(E), TPB, 0, stream>>>(src, dst, deg_src, deg_dst, E);
  k_maxdeg  <<<grid1d(N), TPB, 0, stream>>>(deg_src, maxbits, N);

  // --- per-node scalars + attention gate ---
  k_nodescalar<<<grid1d(N), TPB, 0, stream>>>(deg_src, deg_dst, maxbits, param1, param2,
                                              deg_W, deg_b, att_W, att_b,
                                              nd, dinv, s1, s2, N, H);

  // --- weight packing (tiny) ---
  k_pack_weight<<<grid1d((long)K1 * H2), TPB, 0, stream>>>(W1, W1t, K1, H2, H2);
  k_pack_weight<<<grid1d((long)H2 * H ), TPB, 0, stream>>>(W2, W2t, H2, H,  H );
  k_pack_weight<<<grid1d((long)H  * Cp), TPB, 0, stream>>>(W3, W3t, H,  C,  Cp);

  // --- layer 1 ---
  k_build_xa<<<grid1d((long)N * K1), TPB, 0, stream>>>(x, nd, s1, deg_W, deg_b, xa, N, F, H);
  k_wmma_gemm_lds<8><<<gemmBlocks, TPB, (size_t)H2 * (K1 + 8) * 2, stream>>>(xa, W1t, h1, N, K1, H2);
  k_selfinit<<<grid1d((long)N * H2),    TPB, 0, stream>>>(h1, b1, dinv, g1, N, H2, H2);
  k_scatter4<<<grid1d(E * (H2 / 4)),    TPB, 0, stream>>>(h1, src, dst, dinv, g1, E, H2 / 4, H2, H2);
  k_post1   <<<grid1d((long)N * H2),    TPB, 0, stream>>>(g1, s2, x1b, x1res, N, H2, H);

  // --- layer 2 ---
  k_wmma_gemm_lds<4><<<gemmBlocks, TPB, (size_t)H * (H2 + 8) * 2, stream>>>(x1b, W2t, h2, N, H2, H);
  k_selfinit<<<grid1d((long)N * H),     TPB, 0, stream>>>(h2, b2, dinv, g2, N, H, H);
  k_scatter4<<<grid1d(E * (H / 4)),     TPB, 0, stream>>>(h2, src, dst, dinv, g2, E, H / 4, H, H);
  k_post2   <<<grid1d((long)N * H),     TPB, 0, stream>>>(g2, x1res, x2b, N, H);

  // --- layer 3 (C padded to Cp columns; pad weights are zero) ---
  k_wmma_gemm_lds<3><<<gemmBlocks, TPB, (size_t)Cp * (H + 8) * 2, stream>>>(x2b, W3t, h3, N, H, Cp);
  k_selfinit<<<grid1d((long)N * C),     TPB, 0, stream>>>(h3, b3, dinv, g3, N, C, Cp);
  k_scatter4<<<grid1d(E * (C / 4)),     TPB, 0, stream>>>(h3, src, dst, dinv, g3, E, C / 4, Cp, C);

  // --- log-softmax ---
  k_logsoftmax<<<grid1d(N), TPB, 0, stream>>>(g3, (float*)d_out, N, C);
}
